// GRUClassifier_15736760172573
// MI455X (gfx1250) — matrix operand order repaired and checked by reading, not compiled, not run on hardware
//
#include <hip/hip_runtime.h>

#define BDIM   256   // threads per block (8 waves of 32)
#define NWAVES 8
#define BT     64    // batch rows per workgroup
#define HD     256   // hidden size H
#define NG     768   // 3*H gate rows
#define TSTEPS 512   // sequence length T
#define DIN0   128   // layer-0 input size D
#define NB     256   // batch size B

typedef __attribute__((ext_vector_type(16))) __bf16 v16bf;
typedef __attribute__((ext_vector_type(8)))  __bf16 v8bf;
typedef __attribute__((ext_vector_type(8)))  float  v8f;

union FragBF {
  v16bf v;
  v8bf  h[2];
};

__device__ __forceinline__ float sigmoidf_(float x) {
  return 1.0f / (1.0f + __expf(-x));
}

// ---------------------------------------------------------------------------
// One GRU cell step for a BT x HD state tile held in LDS.
// Each wave processes two 16-wide gate-column slices (nt = wv, wv+8); for a
// slice it computes all four 16-row M-tiles with SHARED B-fragments: each
// weight fragment load (global/L2) feeds 12 WMMAs (4 m-tiles x 3 gates)
// -> 4x less L2 weight traffic than a per-tile load scheme.
// A-fragments come from LDS. WMMA: v_wmma_f32_16x16x32_bf16, f32 accumulate.
// ---------------------------------------------------------------------------
template<int DIN>
__device__ __forceinline__ void gru_cell_step(
    const __bf16* __restrict__ xin,            // LDS input tile [BT][DIN]
    const __bf16 (* __restrict__ hcur)[HD],    // LDS state  [BT][HD]
    __bf16 (* __restrict__ hnxt)[HD],          // LDS state out [BT][HD]
    const __bf16* __restrict__ wih,            // [NG][DIN] bf16
    const __bf16* __restrict__ whh,            // [NG][HD]  bf16
    const float* __restrict__ bih,
    const float* __restrict__ bhh,
    int lane, int wv,
    float* __restrict__ lastOut, int bBase)    // optional f32 h_new dump
{
  const int mHalf = lane >> 4;      // 0 or 1 (lane half)
  const int nIdx  = lane & 15;      // column within 16x16 tile

  for (int nt = wv; nt < 16; nt += NWAVES) {
    const int n0    = nt << 4;        // this slice's 16 gate columns
    const int colR  = n0 + nIdx;      // r-gate column
    const int colZ  = colR + HD;      // z-gate column
    const int colN  = colR + 2 * HD;  // n-gate column

    v8f accR[4], accZ[4], accGN[4], accHN[4];
    const float bR0 = bih[colR] + bhh[colR];
    const float bZ0 = bih[colZ] + bhh[colZ];
    const float bGN = bih[colN];
    const float bHN = bhh[colN];
#pragma unroll
    for (int mt = 0; mt < 4; ++mt) {
#pragma unroll
      for (int j = 0; j < 8; ++j) {
        accR[mt][j] = bR0; accZ[mt][j] = bZ0; accGN[mt][j] = bGN; accHN[mt][j] = bHN;
      }
    }

    // ---- input contribution: gi = xin @ Wih^T ----
#pragma unroll 2
    for (int k = 0; k < DIN; k += 32) {
      FragBF bR, bZ, bN;
      const int kb = k + mHalf * 16;      // B: lanes 0-15 K 0-15, lanes 16-31 K 16-31
      bR.h[0] = *(const v8bf*)(wih + (size_t)colR * DIN + kb);
      bR.h[1] = *(const v8bf*)(wih + (size_t)colR * DIN + kb + 8);
      bZ.h[0] = *(const v8bf*)(wih + (size_t)colZ * DIN + kb);
      bZ.h[1] = *(const v8bf*)(wih + (size_t)colZ * DIN + kb + 8);
      bN.h[0] = *(const v8bf*)(wih + (size_t)colN * DIN + kb);
      bN.h[1] = *(const v8bf*)(wih + (size_t)colN * DIN + kb + 8);
      const int ka = k + mHalf * 16;      // A: lanes 0-15 K 0-15, lanes 16-31 K 16-31
#pragma unroll
      for (int mt = 0; mt < 4; ++mt) {
        FragBF a;
        const int aRow = mt * 16 + nIdx;
        a.h[0] = *(const v8bf*)(xin + aRow * DIN + ka);
        a.h[1] = *(const v8bf*)(xin + aRow * DIN + ka + 8);
        accR[mt]  = __builtin_amdgcn_wmma_f32_16x16x32_bf16(false, a.v, false, bR.v, (short)0, accR[mt],  false, false);
        accZ[mt]  = __builtin_amdgcn_wmma_f32_16x16x32_bf16(false, a.v, false, bZ.v, (short)0, accZ[mt],  false, false);
        accGN[mt] = __builtin_amdgcn_wmma_f32_16x16x32_bf16(false, a.v, false, bN.v, (short)0, accGN[mt], false, false);
      }
    }

    // ---- hidden contribution: gh = h_cur @ Whh^T ----
#pragma unroll 2
    for (int k = 0; k < HD; k += 32) {
      FragBF bR, bZ, bN;
      const int kb = k + mHalf * 16;
      bR.h[0] = *(const v8bf*)(whh + (size_t)colR * HD + kb);
      bR.h[1] = *(const v8bf*)(whh + (size_t)colR * HD + kb + 8);
      bZ.h[0] = *(const v8bf*)(whh + (size_t)colZ * HD + kb);
      bZ.h[1] = *(const v8bf*)(whh + (size_t)colZ * HD + kb + 8);
      bN.h[0] = *(const v8bf*)(whh + (size_t)colN * HD + kb);
      bN.h[1] = *(const v8bf*)(whh + (size_t)colN * HD + kb + 8);
      const int ka = k + mHalf * 16;
#pragma unroll
      for (int mt = 0; mt < 4; ++mt) {
        FragBF a;
        const int aRow = mt * 16 + nIdx;
        a.h[0] = *(const v8bf*)(&hcur[aRow][ka]);
        a.h[1] = *(const v8bf*)(&hcur[aRow][ka + 8]);
        accR[mt]  = __builtin_amdgcn_wmma_f32_16x16x32_bf16(false, a.v, false, bR.v, (short)0, accR[mt],  false, false);
        accZ[mt]  = __builtin_amdgcn_wmma_f32_16x16x32_bf16(false, a.v, false, bZ.v, (short)0, accZ[mt],  false, false);
        accHN[mt] = __builtin_amdgcn_wmma_f32_16x16x32_bf16(false, a.v, false, bN.v, (short)0, accHN[mt], false, false);
      }
    }

    // ---- gates + state update (C/D layout: VGPR j -> M = j + 8*laneHalf) ----
#pragma unroll
    for (int mt = 0; mt < 4; ++mt) {
#pragma unroll
      for (int j = 0; j < 8; ++j) {
        const int m = mt * 16 + mHalf * 8 + j;
        const float hOld = (float)hcur[m][colR];
        const float r = sigmoidf_(accR[mt][j]);
        const float z = sigmoidf_(accZ[mt][j]);
        const float n = tanhf(accGN[mt][j] + r * accHN[mt][j]);
        const float hN = (1.0f - z) * n + z * hOld;
        hnxt[m][colR] = (__bf16)hN;
        if (lastOut) lastOut[(size_t)(bBase + m) * HD + colR] = hN;
      }
    }
  }
}

// ---------------------------------------------------------------------------
// Fused 2-layer GRU scan. grid.x = NB/BT workgroups; each owns a batch tile.
// h-state for both layers stays double-buffered in LDS for all 512 steps;
// layer-1 consumes layer-0's step-t output directly from LDS (no HBM traffic).
// ---------------------------------------------------------------------------
__global__ __launch_bounds__(BDIM)
void gru2_scan_kernel(const float* __restrict__ x,        // [B][T][DIN0] f32
                      const __bf16* __restrict__ wih0, const __bf16* __restrict__ whh0,
                      const float* __restrict__ bih0,  const float* __restrict__ bhh0,
                      const __bf16* __restrict__ wih1, const __bf16* __restrict__ whh1,
                      const float* __restrict__ bih1,  const float* __restrict__ bhh1,
                      float* __restrict__ hlast)          // [B][HD] f32
{
  __shared__ __bf16 xs[BT][DIN0];        // 16 KB
  __shared__ __bf16 h0[2][BT][HD];       // 64 KB
  __shared__ __bf16 h1[2][BT][HD];       // 64 KB

  const int tid   = threadIdx.x;
  const int lane  = tid & 31;
  const int wv    = tid >> 5;
  const int bBase = blockIdx.x * BT;

  for (int i = tid; i < 2 * BT * HD; i += BDIM) {
    (&h0[0][0][0])[i] = (__bf16)0.0f;
    (&h1[0][0][0])[i] = (__bf16)0.0f;
  }
  __syncthreads();

  int cur = 0;
  for (int t = 0; t < TSTEPS; ++t) {
    // stage x_t tile: f32 global -> bf16 LDS (coalesced float4 reads)
    for (int i = tid * 4; i < BT * DIN0; i += BDIM * 4) {
      const int r = i >> 7;               // / DIN0
      const int c = i & (DIN0 - 1);
      const float4 v = *(const float4*)(x + ((size_t)(bBase + r) * TSTEPS + t) * DIN0 + c);
      xs[r][c + 0] = (__bf16)v.x;
      xs[r][c + 1] = (__bf16)v.y;
      xs[r][c + 2] = (__bf16)v.z;
      xs[r][c + 3] = (__bf16)v.w;
    }
    __syncthreads();

    // layer 0 cell: x_t, h0[cur] -> h0[cur^1]
    gru_cell_step<DIN0>(&xs[0][0], h0[cur], h0[cur ^ 1], wih0, whh0, bih0, bhh0,
                        lane, wv, nullptr, bBase);
    __syncthreads();

    // layer 1 cell: fresh h0 output is the input; h1[cur] -> h1[cur^1]
    gru_cell_step<HD>(&h0[cur ^ 1][0][0], h1[cur], h1[cur ^ 1], wih1, whh1, bih1, bhh1,
                      lane, wv, (t == TSTEPS - 1) ? hlast : nullptr, bBase);
    __syncthreads();

    cur ^= 1;
  }
}

// ---------------------------------------------------------------------------
__global__ void f32_to_bf16_kernel(const float* __restrict__ in,
                                   __bf16* __restrict__ out, int n) {
  const int i = blockIdx.x * blockDim.x + threadIdx.x;
  if (i < n) out[i] = (__bf16)in[i];
}

__global__ void fc_kernel(const float* __restrict__ last,   // [B][HD]
                          const float* __restrict__ Wfc,    // [C][HD]
                          const float* __restrict__ bfc,    // [C]
                          float* __restrict__ out) {        // [B][C]
  const int idx = blockIdx.x * blockDim.x + threadIdx.x;
  if (idx >= NB * 10) return;
  const int b = idx / 10;
  const int c = idx % 10;
  float s = bfc[c];
  for (int h = 0; h < HD; ++h) s += last[(size_t)b * HD + h] * Wfc[(size_t)c * HD + h];
  out[idx] = s;
}

// ---------------------------------------------------------------------------
extern "C" void kernel_launch(void* const* d_in, const int* in_sizes, int n_in,
                              void* d_out, int out_size, void* d_ws, size_t ws_size,
                              hipStream_t stream) {
  (void)in_sizes; (void)n_in; (void)out_size; (void)ws_size;

  const float* x    = (const float*)d_in[0];
  const float* Wih0 = (const float*)d_in[1];
  const float* Whh0 = (const float*)d_in[2];
  const float* bih0 = (const float*)d_in[3];
  const float* bhh0 = (const float*)d_in[4];
  const float* Wih1 = (const float*)d_in[5];
  const float* Whh1 = (const float*)d_in[6];
  const float* bih1 = (const float*)d_in[7];
  const float* bhh1 = (const float*)d_in[8];
  const float* Wfc  = (const float*)d_in[9];
  const float* bfc  = (const float*)d_in[10];
  float* out = (float*)d_out;

  // carve workspace (256B aligned chunks)
  char* p = (char*)d_ws;
  auto carve = [&](size_t bytes) -> char* {
    char* r = p;
    p += (bytes + 255) & ~(size_t)255;
    return r;
  };
  __bf16* wih0_bf = (__bf16*)carve((size_t)NG * DIN0 * sizeof(__bf16));
  __bf16* whh0_bf = (__bf16*)carve((size_t)NG * HD   * sizeof(__bf16));
  __bf16* wih1_bf = (__bf16*)carve((size_t)NG * HD   * sizeof(__bf16));
  __bf16* whh1_bf = (__bf16*)carve((size_t)NG * HD   * sizeof(__bf16));
  float*  hlast   = (float*)carve((size_t)NB * HD    * sizeof(float));

  // weight conversions f32 -> bf16 (tiny, L2-resident afterwards)
  int n;
  n = NG * DIN0; f32_to_bf16_kernel<<<(n + 255) / 256, 256, 0, stream>>>(Wih0, wih0_bf, n);
  n = NG * HD;   f32_to_bf16_kernel<<<(n + 255) / 256, 256, 0, stream>>>(Whh0, whh0_bf, n);
  n = NG * HD;   f32_to_bf16_kernel<<<(n + 255) / 256, 256, 0, stream>>>(Wih1, wih1_bf, n);
  n = NG * HD;   f32_to_bf16_kernel<<<(n + 255) / 256, 256, 0, stream>>>(Whh1, whh1_bf, n);

  // fused 2-layer recurrent scan (the serial bottleneck, WMMA-driven)
  gru2_scan_kernel<<<NB / BT, BDIM, 0, stream>>>(
      x, wih0_bf, whh0_bf, bih0, bhh0, wih1_bf, whh1_bf, bih1, bhh1, hlast);

  // classifier head
  fc_kernel<<<(NB * 10 + 255) / 256, 256, 0, stream>>>(hlast, Wfc, bfc, out);
}
